// DynConv2d_32650341384593
// MI455X (gfx1250) — matrix operand order repaired and checked by reading, not compiled, not run on hardware
//
#include <hip/hip_runtime.h>

typedef __attribute__((ext_vector_type(16))) _Float16 v16h;
typedef __attribute__((ext_vector_type(8)))  _Float16 v8h;
typedef __attribute__((ext_vector_type(8)))  float    v8f;

#define BATCH 8
#define CH    64
#define NPT   4096
#define KNN   16
#define OUTC  64

union HalfPack { v16h v; v8h h[2]; };

static __device__ __forceinline__ v8f wmma_f16(v16h a, v16h b, v8f c) {
  // 8 args: (neg_a, A, neg_b, B, c_mod, C, reuse_a, reuse_b)
  return __builtin_amdgcn_wmma_f32_16x16x32_f16(false, a, false, b, (short)0, c,
                                                false, false);
}

// ---------------------------------------------------------------------------
// Kernel 1: transpose x (B,C,N) -> xt (B,N,C) in f16, plus squared norms (f32)
// ---------------------------------------------------------------------------
__global__ __launch_bounds__(256) void prep_xt(const float* __restrict__ x,
                                               _Float16* __restrict__ xt,
                                               float* __restrict__ sq) {
  int p = blockIdx.x * 256 + threadIdx.x;     // 0 .. B*N-1
  int b = p >> 12;
  int n = p & (NPT - 1);
  const float* xp = x + (size_t)b * CH * NPT + n;
  _Float16* op = xt + (size_t)p * CH;
  float s = 0.f;
  for (int c = 0; c < CH; ++c) {
    float v = xp[(size_t)c * NPT];
    op[c] = (_Float16)v;
    s += v * v;
  }
  sq[p] = s;
}

// ---------------------------------------------------------------------------
// Kernel 2: weights f32 -> f16 (both matrices, each OUTC x 2C row-major)
// ---------------------------------------------------------------------------
__global__ __launch_bounds__(256) void prep_w(const float* __restrict__ We,
                                              const float* __restrict__ Wa,
                                              _Float16* __restrict__ we,
                                              _Float16* __restrict__ wa) {
  int i = blockIdx.x * 256 + threadIdx.x;
  if (i < OUTC * 2 * CH) {
    we[i] = (_Float16)We[i];
    wa[i] = (_Float16)Wa[i];
  }
}

// ---------------------------------------------------------------------------
// Kernel 3: fused distance GEMM (WMMA) + per-row top-16 selection.
// Grid: (B, N/256). Block: 256 threads = 8 INDEPENDENT waves, no barriers.
// Each wave owns 32 rows (two 16-row A tiles); per 32-col stripe it does
// 8 WMMAs into a wave-private LDS slab, then its 32 lanes scan one row each.
// Score folded into the accumulator: D = G - 0.5*sq[col] seeded via C, so
// the WMMA output IS the selection score (maximize <=> min distance).
// ---------------------------------------------------------------------------
__global__ __launch_bounds__(256) void knn_kernel(const _Float16* __restrict__ xt,
                                                  const float* __restrict__ sq,
                                                  int* __restrict__ nn_idx) {
  __shared__ float g_lds[8][32][34];   // wave-private slabs, ~34 KB total

  const int b    = blockIdx.x;
  const int rb   = blockIdx.y;       // 256-row block
  const int tid  = threadIdx.x;
  const int wave = tid >> 5;
  const int lane = tid & 31;
  const int hs   = lane >> 4;        // lane half (ISA A/B layout split)
  const int lm   = lane & 15;
  const int rowBase = rb * 256 + 32 * wave;   // this wave's 32 rows
  const _Float16* xb = xt + (size_t)b * NPT * CH;
  const float*    sqb = sq + (size_t)b * NPT;

  // Preload A fragments for the wave's two 16-row tiles.
  // A frag, contiguous K order chosen here (B matches): lane(0-15)=row M, halves0-15 = K kb+16*hs..+15
  // (lane 16-31 same rows, other K half).
  v16h afrag[2][2];
  for (int t = 0; t < 2; ++t) {
    int m = rowBase + 16 * t + lm;
    const _Float16* rp = xb + (size_t)m * CH;
    for (int kc = 0; kc < 2; ++kc) {
      HalfPack u;
      u.h[0] = *(const v8h*)(rp + 32 * kc + 16 * hs);
      u.h[1] = *(const v8h*)(rp + 32 * kc + 16 * hs + 8);
      afrag[t][kc] = u.v;
    }
  }

  // Per-thread top-16 list, sorted DESCENDING by score (= -dist/2 + const).
  float dlist[KNN];
  int   ilist[KNN];
  for (int i = 0; i < KNN; ++i) { dlist[i] = -3.0e38f; ilist[i] = 0; }

  for (int c0 = 0; c0 < NPT; c0 += 32) {
    // Two 16-col tiles per stripe; all loads issued before the WMMAs.
    const _Float16* cp0 = xb + (size_t)(c0 + lm) * CH;
    const _Float16* cp1 = xb + (size_t)(c0 + 16 + lm) * CH;
    v16h b00 = *(const v16h*)(cp0 + 16 * hs);
    v16h b01 = *(const v16h*)(cp0 + 32 + 16 * hs);
    v16h b10 = *(const v16h*)(cp1 + 16 * hs);
    v16h b11 = *(const v16h*)(cp1 + 32 + 16 * hs);
    float sq0 = -0.5f * sqb[c0 + lm];
    float sq1 = -0.5f * sqb[c0 + 16 + lm];

    // Prefetch next stripe's 32 B rows (global_prefetch_b8), one per lane.
    {
      int nc = c0 + 32;
      if (nc >= NPT) nc = 0;
      __builtin_prefetch((const void*)(xb + (size_t)(nc + lane) * CH), 0, 3);
    }

    v8f a0c0, a0c1, a1c0, a1c1;   // [row tile][col tile]
    for (int r = 0; r < 8; ++r) {
      a0c0[r] = sq0; a1c0[r] = sq0;
      a0c1[r] = sq1; a1c1[r] = sq1;
    }
    a0c0 = wmma_f16(afrag[0][0], b00, a0c0);
    a0c0 = wmma_f16(afrag[0][1], b01, a0c0);
    a1c0 = wmma_f16(afrag[1][0], b00, a1c0);
    a1c0 = wmma_f16(afrag[1][1], b01, a1c0);
    a0c1 = wmma_f16(afrag[0][0], b10, a0c1);
    a0c1 = wmma_f16(afrag[0][1], b11, a0c1);
    a1c1 = wmma_f16(afrag[1][0], b10, a1c1);
    a1c1 = wmma_f16(afrag[1][1], b11, a1c1);

    // C layout: lane = col (lm), VGPR r -> row 8*hs + r of the 16-row tile.
    for (int r = 0; r < 8; ++r) {
      g_lds[wave][8 * hs + r][lm]           = a0c0[r];
      g_lds[wave][8 * hs + r][16 + lm]      = a0c1[r];
      g_lds[wave][16 + 8 * hs + r][lm]      = a1c0[r];
      g_lds[wave][16 + 8 * hs + r][16 + lm] = a1c1[r];
    }

    // Wave-local hand-off: same-wave LDS ops are in order (DScnt only, no
    // barrier). Lane L scans row L of this wave's slab.
    for (int j = 0; j < 32; ++j) {
      float val = g_lds[wave][lane][j];
      if (val > dlist[KNN - 1]) {
        int pos = KNN - 1;
        while (pos > 0 && dlist[pos - 1] < val) {
          dlist[pos] = dlist[pos - 1];
          ilist[pos] = ilist[pos - 1];
          --pos;
        }
        dlist[pos] = val;
        ilist[pos] = c0 + j;
      }
    }
  }

  int row = rowBase + lane;
  int* op = nn_idx + ((size_t)b * NPT + row) * KNN;
  for (int i = 0; i < KNN; ++i) op[i] = ilist[i];
}

// ---------------------------------------------------------------------------
// Kernel 4: gather neighbors -> feat(16x128) in LDS, two WMMA GEMMs
// (edge & attention logits), softmax over k, attention-weighted sum.
// One wave per point; 8 points per 256-thread block.
// ---------------------------------------------------------------------------
__global__ __launch_bounds__(256) void edge_kernel(const _Float16* __restrict__ xt,
                                                   const int* __restrict__ nn_idx,
                                                   const _Float16* __restrict__ We,
                                                   const _Float16* __restrict__ Wa,
                                                   const float* __restrict__ be,
                                                   const float* __restrict__ ba,
                                                   float* __restrict__ out) {
  __shared__ _Float16 feat[8][16 * 128];   // 32 KB

  const int tid  = threadIdx.x;
  const int wave = tid >> 5;
  const int lane = tid & 31;
  const int p = blockIdx.x * 8 + wave;     // global point
  const int b = p >> 12;
  const int n = p & (NPT - 1);

  const _Float16* xb = xt + (size_t)b * NPT * CH;
  const _Float16* xi = xb + (size_t)n * CH;
  _Float16* f = feat[wave];

  // Build feat rows: 2 lanes per neighbor, 32 channels each.
  {
    int kk = lane >> 1;
    int cb = 32 * (lane & 1);
    int idx = nn_idx[(size_t)p * KNN + kk];
    const _Float16* xj = xb + (size_t)idx * CH;
    for (int c = 0; c < 32; ++c) {
      float vi = (float)xi[cb + c];
      float vj = (float)xj[cb + c];
      f[kk * 128 + cb + c]      = (_Float16)vi;          // x_i
      f[kk * 128 + CH + cb + c] = (_Float16)(vj - vi);   // x_j - x_i
    }
  }
  __syncthreads();

  const int hs = lane >> 4;
  const int lm = lane & 15;

  // A fragments: M = neighbor (16), K = 128 channels in 4 chunks of 32.
  v16h afrag[4];
  for (int kc = 0; kc < 4; ++kc) {
    const _Float16* rp = f + lm * 128 + 32 * kc;
    HalfPack u;
    u.h[0] = *(const v8h*)(rp + 16 * hs);
    u.h[1] = *(const v8h*)(rp + 16 * hs + 8);
    afrag[kc] = u.v;
  }

  // Four 16-wide output column tiles covering O=64.
  for (int ot = 0; ot < 4; ++ot) {
    int o = 16 * ot + lm;                // this lane's output column
    float biasE = be[o];
    float biasA = ba[o];
    v8f cE, cA;
    for (int r = 0; r < 8; ++r) { cE[r] = biasE; cA[r] = biasA; }

    for (int kc = 0; kc < 4; ++kc) {
      // B frag: lane = output col, 16 contiguous K halves of W[o][*].
      v16h bE = *(const v16h*)(We + (size_t)o * 128 + 32 * kc + 16 * hs);
      v16h bA = *(const v16h*)(Wa + (size_t)o * 128 + 32 * kc + 16 * hs);
      cE = wmma_f16(afrag[kc], bE, cE);
      cA = wmma_f16(afrag[kc], bA, cA);
    }

    // Softmax over k (row/M dim): 8 rows in-lane + 8 rows in lane^16.
    float vmax = cA[0];
    for (int r = 1; r < 8; ++r) vmax = fmaxf(vmax, cA[r]);
    vmax = fmaxf(vmax, __shfl_xor(vmax, 16, 32));

    float s = 0.f, accn = 0.f;
    for (int r = 0; r < 8; ++r) {
      float e = __expf(cA[r] - vmax);
      s += e;
      accn += cE[r] * e;
    }
    s    += __shfl_xor(s, 16, 32);
    accn += __shfl_xor(accn, 16, 32);

    if (lane < 16) {
      out[((size_t)b * OUTC + (16 * ot + lane)) * NPT + n] = accn / s;
    }
  }
}

// ---------------------------------------------------------------------------
extern "C" void kernel_launch(void* const* d_in, const int* in_sizes, int n_in,
                              void* d_out, int out_size, void* d_ws, size_t ws_size,
                              hipStream_t stream) {
  (void)in_sizes; (void)n_in; (void)out_size; (void)ws_size;
  const float* x      = (const float*)d_in[0];
  const float* W_edge = (const float*)d_in[1];
  const float* b_edge = (const float*)d_in[2];
  const float* W_att  = (const float*)d_in[3];
  const float* b_att  = (const float*)d_in[4];
  // d_in[5] = k (==16), hardcoded.

  char* ws = (char*)d_ws;
  _Float16* xt = (_Float16*)(ws);                 // B*N*C f16      = 4 MB
  float*    sq = (float*)(ws + 4194304);          // B*N f32        = 128 KB
  _Float16* we = (_Float16*)(ws + 4325376);       // 64*128 f16     = 16 KB
  _Float16* wa = (_Float16*)(ws + 4341760);       // 64*128 f16     = 16 KB
  int*      nn = (int*)(ws + 4358144);            // B*N*16 i32     = 2 MB
  float*    out = (float*)d_out;

  prep_xt<<<dim3(BATCH * NPT / 256), dim3(256), 0, stream>>>(x, xt, sq);
  prep_w<<<dim3((OUTC * 2 * CH + 255) / 256), dim3(256), 0, stream>>>(W_edge, W_att, we, wa);
  knn_kernel<<<dim3(BATCH, NPT / 256), dim3(256), 0, stream>>>(xt, sq, nn);
  edge_kernel<<<dim3(BATCH * NPT / 8), dim3(256), 0, stream>>>(xt, nn, we, wa,
                                                               b_edge, b_att, out);
}
